// PrototypeAdapter_89292370084150
// MI455X (gfx1250) — matrix operand order repaired and checked by reading, not compiled, not run on hardware
//
#include <hip/hip_runtime.h>
#include <hip/hip_bf16.h>

// CDNA5 (gfx1250) wave32 WMMA types
typedef __attribute__((ext_vector_type(16))) __bf16 v16bf;
typedef __attribute__((ext_vector_type(8)))  float  v8f;

#define HID    1024   // hidden
#define BOT    128    // bottleneck
#define NCLUST 16
#define MT     32     // tokens per block (two 16-row WMMA M-tiles)
#define THREADS 128   // 4 wave32

// ---------- helpers ----------
__device__ __forceinline__ unsigned short f2bf(float a) {
  unsigned int ua = __float_as_uint(a);
  return (unsigned short)((ua + 0x7FFFu + ((ua >> 16) & 1u)) >> 16);  // RNE
}
__device__ __forceinline__ unsigned int pack2bf(float a, float b) {
  return (unsigned int)f2bf(a) | ((unsigned int)f2bf(b) << 16);
}
// Branchless tanh-form GELU: 0.5x(1+tanh(u)) == x*sigmoid(2u); |err|~3e-4 abs,
// below the bf16 quantization noise on mid. One v_exp_f32, no EXEC divergence.
__device__ __forceinline__ float gelu_fast(float x) {
  float u = 0.7978845608028654f * (x + 0.044715f * x * x * x);
  return x / (1.0f + __expf(-2.0f * u));
}
// 16-bit A-matrix 16x32 per-lane K offset (ISA 7.12.2): V0:K0,1 .. V3:K6,7 ; V4..7:K16..23 ; +8 for lanes16-31
__device__ __forceinline__ int kaoff(int i, int hf) {
  return ((i < 4) ? (2 * i) : (16 + 2 * (i - 4))) + 8 * hf;
}
// Build A fragment from fp32 rows staged in LDS.
// fp32->bf16 truncation + pair pack in ONE v_perm_b32 per pair.
__device__ __forceinline__ v16bf afrag_f32(const float* rowp, int hf, int k0) {
  union { v16bf v; unsigned int u[8]; } r;
#pragma unroll
  for (int i = 0; i < 8; ++i) {
    const float2 f = *(const float2*)(rowp + k0 + kaoff(i, hf));
    // dst[15:0]=hi16(f.x), dst[31:16]=hi16(f.y)
    r.u[i] = __builtin_amdgcn_perm(__float_as_uint(f.y), __float_as_uint(f.x), 0x07060302u);
  }
  return r.v;
}
// Build A fragment from bf16 rows in LDS (pairs are 32-bit aligned: K offsets are even)
__device__ __forceinline__ v16bf afrag_bf16(const unsigned short* rowp, int hf, int k0) {
  union { v16bf v; unsigned int u[8]; } r;
#pragma unroll
  for (int i = 0; i < 8; ++i)
    r.u[i] = *(const unsigned int*)(rowp + k0 + kaoff(i, hf));
  return r.v;
}
// B fragment: lane%16 = N column, 16 K elements in the same order as A (kaoff), strided by ld
__device__ __forceinline__ v16bf bfrag(const unsigned short* p, int ld, int hf) {
  union { v16bf v; unsigned short e[16]; } r;
#pragma unroll
  for (int i = 0; i < 8; ++i) {
    const int kk = kaoff(i, hf);
    r.e[2 * i]     = p[(size_t)kk * ld];
    r.e[2 * i + 1] = p[(size_t)(kk + 1) * ld];
  }
  return r.v;
}

// ---------- prepass kernels ----------
// meta layout (ints): [0..15]=cnt  [16..31]=cursor  [32..47]=base  [48..64]=tileBase(17)
__global__ void zero_meta_kernel(int* __restrict__ meta) {
  if (threadIdx.x < 32) meta[threadIdx.x] = 0;
}
__global__ void count_kernel(const int* __restrict__ ids, int n, int* __restrict__ meta) {
  int i = blockIdx.x * blockDim.x + threadIdx.x;
  if (i < n) atomicAdd(&meta[ids[i]], 1);
}
__global__ void prefix_kernel(int* __restrict__ meta) {
  int acc = 0, tacc = 0;
  for (int c = 0; c < NCLUST; ++c) {
    meta[32 + c] = acc;
    meta[48 + c] = tacc;
    int cnt = meta[c];
    acc += cnt;
    tacc += (cnt + MT - 1) / MT;
  }
  meta[48 + NCLUST] = tacc;  // total tiles
}
__global__ void scatter_kernel(const int* __restrict__ ids, int n,
                               int* __restrict__ meta, int* __restrict__ sorted_idx) {
  int i = blockIdx.x * blockDim.x + threadIdx.x;
  if (i < n) {
    int c = ids[i];
    int pos = meta[32 + c] + atomicAdd(&meta[16 + c], 1);
    sorted_idx[pos] = i;
  }
}
// fp32 -> packed bf16 pairs (RNE; runs once, off the critical path)
__global__ void cvt_bf16_kernel(const float* __restrict__ src, unsigned int* __restrict__ dst, int npairs) {
  int i = blockIdx.x * blockDim.x + threadIdx.x;
  if (i < npairs) {
    float2 f = ((const float2*)src)[i];
    dst[i] = pack2bf(f.x, f.y);
  }
}

// ---------- main tile kernel ----------
__global__ __launch_bounds__(THREADS)
void adapter_tiles_kernel(const float* __restrict__ hsrc,
                          const float* __restrict__ b1,
                          const float* __restrict__ b2,
                          const unsigned short* __restrict__ W1bf,
                          const unsigned short* __restrict__ W2bf,
                          const int* __restrict__ meta,
                          const int* __restrict__ sorted_idx,
                          float* __restrict__ out)
{
  __shared__ float sX[MT][HID + 4];             // 131.6 KB fp32 token rows (residual + A source)
  __shared__ unsigned short sMid[MT][BOT + 8];  // 8.7 KB bf16 bottleneck activations
  __shared__ int sTok[MT];

  const int b = blockIdx.x;
  if (b >= meta[48 + NCLUST]) return;           // uniform early-exit (EXEC stays all-ones for WMMA)

  int c = 0;
#pragma unroll
  for (int i = 1; i < NCLUST; ++i) if (b >= meta[48 + i]) c = i;
  const int start = (b - meta[48 + c]) * MT;
  const int cbase = meta[32 + c];
  int valid = meta[c] - start;
  valid = valid > MT ? MT : valid;

  const int tid = threadIdx.x;
  if (tid < MT) sTok[tid] = sorted_idx[cbase + ((tid < valid) ? (start + tid) : 0)];
  __syncthreads();

  // Stage MT fp32 rows of h into LDS (coalesced float4 stream; h read exactly once)
  {
    const int row = tid >> 2;
    const int col0 = (tid & 3) * 256;
    const float* src = hsrc + (size_t)sTok[row] * HID + col0;
    float* dst = &sX[row][col0];
#pragma unroll
    for (int i = 0; i < 64; ++i)
      *(float4*)(dst + 4 * i) = *(const float4*)(src + 4 * i);
  }
  __syncthreads();

  const int lane = tid & 31;
  const int wv   = tid >> 5;
  const int mr   = lane & 15;
  const int hf   = lane >> 4;

  // ===== GEMM1: mid = gelu(X @ W1c + b1c)   X:32x1024  W1c:1024x128 =====
  const unsigned short* w1c = W1bf + (size_t)c * HID * BOT;
  const int n0 = wv * 32;                       // each wave owns two 16-wide N tiles
  v8f acc00 = {}, acc01 = {}, acc10 = {}, acc11 = {};
  for (int k0 = 0; k0 < HID; k0 += 32) {
    const v16bf a0 = afrag_f32(&sX[0][0]  + mr * (HID + 4), hf, k0);
    const v16bf a1 = afrag_f32(&sX[16][0] + mr * (HID + 4), hf, k0);
    const unsigned short* bp = w1c + (size_t)k0 * BOT + n0 + mr;
    const v16bf bA = bfrag(bp, BOT, hf);
    const v16bf bB = bfrag(bp + 16, BOT, hf);
    acc00 = __builtin_amdgcn_wmma_f32_16x16x32_bf16(false, a0, false, bA, (short)0, acc00, false, false);
    acc01 = __builtin_amdgcn_wmma_f32_16x16x32_bf16(false, a0, false, bB, (short)0, acc01, false, false);
    acc10 = __builtin_amdgcn_wmma_f32_16x16x32_bf16(false, a1, false, bA, (short)0, acc10, false, false);
    acc11 = __builtin_amdgcn_wmma_f32_16x16x32_bf16(false, a1, false, bB, (short)0, acc11, false, false);
  }

  // bias + GELU -> bf16 mid in LDS (C/D layout: N=lane%16, M=r+8*half)
  {
    const float* b1c = b1 + c * BOT;
    const float bias0 = b1c[n0 + mr];
    const float bias1 = b1c[n0 + 16 + mr];
#pragma unroll
    for (int r = 0; r < 8; ++r) {
      const int m = r + 8 * hf;
      sMid[m][n0 + mr]           = f2bf(gelu_fast(acc00[r] + bias0));
      sMid[m][n0 + 16 + mr]      = f2bf(gelu_fast(acc01[r] + bias1));
      sMid[16 + m][n0 + mr]      = f2bf(gelu_fast(acc10[r] + bias0));
      sMid[16 + m][n0 + 16 + mr] = f2bf(gelu_fast(acc11[r] + bias1));
    }
  }
  __syncthreads();

  // ===== GEMM2: out = h + mid @ W2c + b2c   mid:32x128  W2c:128x1024 =====
  // hoist all mid A-fragments (K=0..127 -> 4 steps, 2 M-tiles)
  v16bf am0[4], am1[4];
#pragma unroll
  for (int s = 0; s < 4; ++s) {
    am0[s] = afrag_bf16(&sMid[0][0]  + mr * (BOT + 8), hf, 32 * s);
    am1[s] = afrag_bf16(&sMid[16][0] + mr * (BOT + 8), hf, 32 * s);
  }

  const unsigned short* w2c = W2bf + (size_t)c * BOT * HID;
  const float* b2c = b2 + c * HID;
  for (int j = 0; j < 16; ++j) {                // each wave covers 16 N-tiles -> 4 waves span H=1024
    const int nb = wv * 256 + j * 16;
    v8f d0 = {}, d1 = {};
#pragma unroll
    for (int s = 0; s < 4; ++s) {
      const v16bf bb = bfrag(w2c + (size_t)(32 * s) * HID + nb + mr, HID, hf);
      d0 = __builtin_amdgcn_wmma_f32_16x16x32_bf16(false, am0[s], false, bb, (short)0, d0, false, false);
      d1 = __builtin_amdgcn_wmma_f32_16x16x32_bf16(false, am1[s], false, bb, (short)0, d1, false, false);
    }
    const int nn = nb + mr;
    const float bias = b2c[nn];
#pragma unroll
    for (int r = 0; r < 8; ++r) {
      const int m = r + 8 * hf;
      if (m < valid)      out[(size_t)sTok[m] * HID + nn]      = sX[m][nn]      + d0[r] + bias;
      if (16 + m < valid) out[(size_t)sTok[16 + m] * HID + nn] = sX[16 + m][nn] + d1[r] + bias;
    }
  }
}

// ---------- launcher ----------
extern "C" void kernel_launch(void* const* d_in, const int* in_sizes, int n_in,
                              void* d_out, int out_size, void* d_ws, size_t ws_size,
                              hipStream_t stream) {
  const float* h   = (const float*)d_in[0];
  const int*   ids = (const int*)d_in[1];
  const float* W1  = (const float*)d_in[2];
  const float* b1  = (const float*)d_in[3];
  const float* W2  = (const float*)d_in[4];
  const float* b2  = (const float*)d_in[5];
  float* out = (float*)d_out;
  const int N = in_sizes[1];  // token count

  // workspace layout
  char* ws = (char*)d_ws;
  int* meta       = (int*)ws;                                  // 65 ints (reserve 256 B)
  int* sorted_idx = (int*)(ws + 256);                          // N ints
  unsigned short* W1bf = (unsigned short*)(ws + 256 + (size_t)N * 4);
  unsigned short* W2bf = W1bf + (size_t)NCLUST * HID * BOT;

  const int wpairs = NCLUST * HID * BOT / 2;

  zero_meta_kernel<<<1, 64, 0, stream>>>(meta);
  cvt_bf16_kernel<<<(wpairs + 255) / 256, 256, 0, stream>>>(W1, (unsigned int*)W1bf, wpairs);
  cvt_bf16_kernel<<<(wpairs + 255) / 256, 256, 0, stream>>>(W2, (unsigned int*)W2bf, wpairs);
  count_kernel<<<(N + 255) / 256, 256, 0, stream>>>(ids, N, meta);
  prefix_kernel<<<1, 1, 0, stream>>>(meta);
  scatter_kernel<<<(N + 255) / 256, 256, 0, stream>>>(ids, N, meta, sorted_idx);

  const int maxTiles = N / MT + NCLUST;  // upper bound on sum(ceil(cnt_c/MT))
  adapter_tiles_kernel<<<maxTiles, THREADS, 0, stream>>>(h, b1, b2, W1bf, W2bf, meta, sorted_idx, out);
}
